// FCOS_1812476198967
// MI455X (gfx1250) — matrix operand order repaired and checked by reading, not compiled, not run on hardware
//
#include <hip/hip_runtime.h>
#include <hip/hip_bf16.h>
#include <math.h>

// ---------------------------------------------------------------------------
// FCOS head for MI455X (gfx1250).  Conv3x3 trunk layers run as implicit-GEMM
// on the wave32 WMMA path (v_wmma_f32_16x16x32_f16).  K-order is (tap,cin) so
// A/B fragments are contiguous vector loads: A = 2x16B global, B = 2x16B LDS
// (ds_load_b128).  Activations kept f16 end-to-end to halve HBM traffic.
// ---------------------------------------------------------------------------

typedef __attribute__((ext_vector_type(16))) _Float16 v16h;
typedef __attribute__((ext_vector_type(8)))  _Float16 v8h;
typedef __attribute__((ext_vector_type(8)))  float    v8f;

#define CIN   256
#define KTOT  2304        // 9 taps * 256 cin
#define NB    2
#define NCLS  20
#define KLVL  500
#define KALL  2500
#define NDETS 100
#define IMGH  800.0f
#define IMGW  1088.0f
#define LDST  264         // padded LDS row stride (halfs): 16B-aligned, bank-skewed

__device__ __forceinline__ float sigm(float x) { return 1.0f / (1.0f + __expf(-x)); }

// ----------------------------- f32 -> f16 (elementwise) ---------------------
__global__ void cvt_f16(const float* __restrict__ w, _Float16* __restrict__ o, int nElem) {
    int t = blockIdx.x * 256 + threadIdx.x;
    if (t < nElem) o[t] = (_Float16)w[t];
}

// --------------- weight f32 [co][cin][tap] -> f16 [co][tap][cin] ------------
__global__ void cvt_w(const float* __restrict__ w, _Float16* __restrict__ o, int nElem) {
    int t = blockIdx.x * 256 + threadIdx.x;
    if (t >= nElem) return;
    int co  = t / KTOT;
    int k   = t - co * KTOT;        // dest k = tap*256 + cin
    int tap = k >> 8;
    int cin = k & 255;
    o[t] = (_Float16)w[(co * CIN + cin) * 9 + tap];
}

// -------------------- conv3x3 (C=256 -> C=256) via WMMA ---------------------
// grid: (pixTiles, 4 coutTiles, NB); block: 128 threads = 4 waves.
// Workgroup tile: 64 Cout x 64 pixels (8x8 spatial).  Wave: 16 Cout rows x 64
// pixels = 4 v_wmma_f32_16x16x32_f16 accumulators.
__global__ __launch_bounds__(128)
void conv3x3_wmma(const _Float16* __restrict__ X, const _Float16* __restrict__ Wf,
                  const float* __restrict__ bias, float* __restrict__ Y,
                  int H, int Wd) {
    __shared__ _Float16 lds[100 * LDST];         // [pos(10x10)][cin], padded (52.8 KB)

    const int TX   = (Wd + 7) >> 3;
    const int ty   = blockIdx.x / TX;
    const int tx   = blockIdx.x - ty * TX;
    const int coBase = blockIdx.y * 64;
    const int n    = blockIdx.z;
    const int tid  = threadIdx.x;
    const int y0   = ty * 8 - 1, x0 = tx * 8 - 1;
    const _Float16* Xn = X + (size_t)n * CIN * H * Wd;

    // Stage input halo tile (f16) into transposed LDS layout [pos][cin].
    for (int i = tid; i < CIN * 100; i += 128) {
        int cin = i / 100, pos = i - cin * 100;   // consecutive tid -> consecutive pos
        int dy = pos / 10, dx = pos - dy * 10;
        int gy = y0 + dy, gx = x0 + dx;
        _Float16 v = (_Float16)0.0f;
        if (gy >= 0 && gy < H && gx >= 0 && gx < Wd)
            v = Xn[(cin * H + gy) * Wd + gx];
        lds[pos * LDST + cin] = v;
    }
    __syncthreads();

    const int wave = tid >> 5, lane = tid & 31;
    const int hlf  = lane >> 4;                   // lane half selects K sub-range
    const int nl   = lane & 15;
    const int coA  = coBase + wave * 16 + nl;     // A-matrix row for this lane
    const _Float16* wrow = Wf + (size_t)coA * KTOT;

    // Per-lane pixel coords for the 4 N-groups (B-matrix N = lane%16).
    int pyv[4], pxv[4];
    #pragma unroll
    for (int pg = 0; pg < 4; ++pg) {
        int pix = pg * 16 + nl;
        pyv[pg] = pix >> 3;
        pxv[pg] = pix & 7;
    }

    v8f acc[4] = {};
    #pragma unroll 1
    for (int tap = 0; tap < 9; ++tap) {
        const int ky = tap / 3, kx = tap - ky * 3;
        for (int cb = 0; cb < CIN; cb += 32) {
            const int kb = tap * 256 + cb;
            if (kb + 32 < KTOT) __builtin_prefetch(wrow + kb + 32, 0, 0);  // global_prefetch_b8

            // A fragment: 16-bit A 16x32 layout = two contiguous 8-half runs.
            v8h a0 = *(const v8h*)(wrow + kb + (hlf << 3));
            v8h a1 = *(const v8h*)(wrow + kb + 16 + (hlf << 3));
            v16h afr = __builtin_shufflevector(a0, a1,
                0, 1, 2, 3, 4, 5, 6, 7, 8, 9, 10, 11, 12, 13, 14, 15);

            #pragma unroll
            for (int pg = 0; pg < 4; ++pg) {
                // B fragment: lanes 0-15 K=0..7,16..23; 16-31 K=8..15,24..31
                // (same two 8-channel runs as A) at this lane's tap position.
                int base = (pyv[pg] + ky) * 10 + (pxv[pg] + kx);
                const _Float16* bp = lds + base * LDST + cb + (hlf << 3);
                v8h b0 = *(const v8h*)bp;
                v8h b1 = *(const v8h*)(bp + 16);
                v16h bfr = __builtin_shufflevector(b0, b1,
                    0, 1, 2, 3, 4, 5, 6, 7, 8, 9, 10, 11, 12, 13, 14, 15);
                acc[pg] = __builtin_amdgcn_wmma_f32_16x16x32_f16(
                    false, afr, false, bfr, (short)0, acc[pg], false, false);
            }
        }
    }

    // Epilogue: C/D 16x16 layout — lane: N = lane%16 (pixel), M = v + (lane/16)*8
    #pragma unroll
    for (int pg = 0; pg < 4; ++pg) {
        int gy = ty * 8 + pyv[pg], gx = tx * 8 + pxv[pg];
        #pragma unroll
        for (int vv = 0; vv < 8; ++vv) {
            int coOut = coBase + wave * 16 + vv + hlf * 8;
            if (gy < H && gx < Wd)
                Y[(((size_t)n * CIN + coOut) * H + gy) * Wd + gx] = acc[pg][vv] + bias[coOut];
        }
    }
}

// ------------------------------ GroupNorm -----------------------------------
__global__ void gn_stats(const float* __restrict__ Y, float* __restrict__ meanO,
                         float* __restrict__ rstdO, int HW) {
    __shared__ float s1[256], s2[256];
    int b = blockIdx.x;                   // n*16 + g
    int n = b >> 4, g = b & 15;
    const float* base = Y + ((size_t)n * CIN + g * 16) * HW;
    int M = 16 * HW;
    float a = 0.f, q = 0.f;
    for (int i = threadIdx.x; i < M; i += 256) { float v = base[i]; a += v; q += v * v; }
    s1[threadIdx.x] = a; s2[threadIdx.x] = q; __syncthreads();
    for (int st = 128; st > 0; st >>= 1) {
        if (threadIdx.x < st) { s1[threadIdx.x] += s1[threadIdx.x + st]; s2[threadIdx.x] += s2[threadIdx.x + st]; }
        __syncthreads();
    }
    if (threadIdx.x == 0) {
        float m = s1[0] / M;
        float var = s2[0] / M - m * m;
        meanO[b] = m;
        rstdO[b] = rsqrtf(var + 1e-5f);
    }
}

// normalize + ReLU, emit f16 activations for the next WMMA conv
__global__ void gn_norm(const float* __restrict__ Y, const float* __restrict__ meanO,
                        const float* __restrict__ rstdO, const float* __restrict__ gamma,
                        const float* __restrict__ beta, _Float16* __restrict__ O, int HW) {
    int t = blockIdx.x * 256 + threadIdx.x;
    int total = NB * CIN * HW;
    if (t >= total) return;
    int c = (t / HW) & (CIN - 1);
    int n = t / (CIN * HW);
    int b = n * 16 + (c >> 4);
    float v = (Y[t] - meanO[b]) * rstdO[b] * gamma[c] + beta[c];
    O[t] = (_Float16)fmaxf(v, 0.0f);
}

// --------------------- small-Cout direct conv (finals) ----------------------
__global__ void conv3x3_direct(const _Float16* __restrict__ X, const float* __restrict__ Wt,
                               const float* __restrict__ bs, float* __restrict__ O,
                               int Cout, int H, int Wd, int doRelu) {
    int t = blockIdx.x * 256 + threadIdx.x;
    int HW = H * Wd;
    int total = NB * Cout * HW;
    if (t >= total) return;
    int pos = t % HW;
    int co  = (t / HW) % Cout;
    int n   = t / (HW * Cout);
    int y = pos / Wd, x = pos - y * Wd;
    float acc = bs[co];
    const _Float16* Xn = X + (size_t)n * CIN * HW;
    const float* wb = Wt + (size_t)co * CIN * 9;
    for (int ci = 0; ci < CIN; ++ci) {
        const _Float16* xc = Xn + (size_t)ci * HW;
        const float* wc = wb + ci * 9;
        #pragma unroll
        for (int ky = 0; ky < 3; ++ky) {
            int yy = y + ky - 1;
            if (yy < 0 || yy >= H) continue;
            #pragma unroll
            for (int kx = 0; kx < 3; ++kx) {
                int xx = x + kx - 1;
                if (xx < 0 || xx >= Wd) continue;
                acc += (float)xc[yy * Wd + xx] * wc[ky * 3 + kx];
            }
        }
    }
    if (doRelu) acc = fmaxf(acc, 0.0f);
    O[t] = acc;
}

// ---------------- FCOS score = sqrt(sig(cls)*sig(ctr)), flat (loc,cls) ------
__global__ void score_kernel(const float* __restrict__ cls, const float* __restrict__ ctr,
                             float* __restrict__ sc, int HW) {
    int t = blockIdx.x * 256 + threadIdx.x;
    if (t >= NB * HW) return;
    int n = t / HW, loc = t % HW;
    float tc = sigm(ctr[(size_t)n * HW + loc]);
    const float* cn = cls + (size_t)n * NCLS * HW;
    float* sn = sc + (size_t)n * HW * NCLS;
    for (int c = 0; c < NCLS; ++c)
        sn[loc * NCLS + c] = sqrtf(sigm(cn[c * HW + loc]) * tc);
}

// ----------------- per-(image,level) top-500 (iterative argmax) -------------
__global__ __launch_bounds__(1024)
void topk_kernel(float* __restrict__ sc, int M, float* __restrict__ candS,
                 int* __restrict__ candI, int lvlOff) {
    __shared__ float rs[1024];
    __shared__ int   ri[1024];
    int n = blockIdx.x, tid = threadIdx.x;
    float* s = sc + (size_t)n * M;
    for (int t = 0; t < KLVL; ++t) {
        float best = -2.0f; int bi = 0;
        for (int i = tid; i < M; i += 1024) { float v = s[i]; if (v > best) { best = v; bi = i; } }
        rs[tid] = best; ri[tid] = bi; __syncthreads();
        for (int st = 512; st > 0; st >>= 1) {
            if (tid < st && rs[tid + st] > rs[tid]) { rs[tid] = rs[tid + st]; ri[tid] = ri[tid + st]; }
            __syncthreads();
        }
        if (tid == 0) {
            float v = rs[0]; int idx = ri[0];
            candS[n * KALL + lvlOff + t] = (v > 0.1f) ? v : 0.0f;
            candI[n * KALL + lvlOff + t] = idx;
            s[idx] = -1.0f;                       // mark taken (probs >= 0)
        }
        __syncthreads();
    }
}

// -------------------------- per-level box decode ----------------------------
__global__ void decode_kernel(const float* __restrict__ candS, const int* __restrict__ candI,
                              const float* __restrict__ reg, float* __restrict__ boxes,
                              float* __restrict__ scC, int* __restrict__ labs,
                              int Wd, int HW, float strd, int lvlOff) {
    int t = blockIdx.x * 256 + threadIdx.x;
    if (t >= NB * KLVL) return;
    int n = t / KLVL, k = t % KLVL;
    int slot = n * KALL + lvlOff + k;
    int idx = candI[slot];
    int loc = idx / NCLS, lab = idx - loc * NCLS;
    int y = loc / Wd, x = loc - y * Wd;
    float xs = x * strd + 0.5f * strd;
    float ys = y * strd + 0.5f * strd;
    const float* rn = reg + (size_t)n * 4 * HW;
    float lv = rn[0 * HW + loc] * strd, tv = rn[1 * HW + loc] * strd;
    float rv = rn[2 * HW + loc] * strd, bv = rn[3 * HW + loc] * strd;
    boxes[slot * 4 + 0] = fminf(fmaxf(xs - lv, 0.0f), IMGW);
    boxes[slot * 4 + 1] = fminf(fmaxf(ys - tv, 0.0f), IMGH);
    boxes[slot * 4 + 2] = fminf(fmaxf(xs + rv, 0.0f), IMGW);
    boxes[slot * 4 + 3] = fminf(fmaxf(ys + bv, 0.0f), IMGH);
    scC[slot]  = candS[slot];
    labs[slot] = lab;
}

// ------------- per-image: sort 2500, greedy NMS, top-100 output -------------
__global__ __launch_bounds__(1024)
void nms_kernel(const float* __restrict__ boxes, const float* __restrict__ scC,
                const int* __restrict__ labs, float* __restrict__ outF,
                int* __restrict__ outLab) {
    __shared__ float key[4096];
    __shared__ int   sidx[4096];
    __shared__ int   keepA[4096];
    __shared__ float rs[1024];
    __shared__ int   ri[1024];
    __shared__ float bc[4];

    int n = blockIdx.x, tid = threadIdx.x;

    for (int i = tid; i < 4096; i += 1024) {
        key[i]  = (i < KALL) ? scC[n * KALL + i] : -1e30f;
        sidx[i] = i;
    }
    __syncthreads();

    // Bitonic sort descending (key, sidx)
    for (int k = 2; k <= 4096; k <<= 1) {
        for (int j = k >> 1; j > 0; j >>= 1) {
            for (int i = tid; i < 4096; i += 1024) {
                int ixj = i ^ j;
                if (ixj > i) {
                    bool dirDesc = ((i & k) == 0);
                    float a = key[i], c = key[ixj];
                    bool sw = dirDesc ? (a < c) : (a > c);
                    if (sw) {
                        key[i] = c; key[ixj] = a;
                        int t2 = sidx[i]; sidx[i] = sidx[ixj]; sidx[ixj] = t2;
                    }
                }
            }
            __syncthreads();
        }
    }

    // Each thread owns sorted positions p = tid + s*1024 (s<3 covers 2500)
    float obx[3][4];
    #pragma unroll
    for (int s = 0; s < 3; ++s) {
        int p = tid + s * 1024;
        if (p < KALL) {
            int jj = sidx[p];
            float off = (float)labs[n * KALL + jj] * (IMGW + IMGH);
            #pragma unroll
            for (int c = 0; c < 4; ++c) obx[s][c] = boxes[((size_t)n * KALL + jj) * 4 + c] + off;
        } else {
            obx[s][0] = obx[s][1] = obx[s][2] = obx[s][3] = 0.0f;
        }
    }
    for (int p = tid; p < 4096; p += 1024) keepA[p] = (p < KALL) ? 1 : 0;
    __syncthreads();

    // Greedy NMS over sorted order
    for (int i = 0; i < KALL; ++i) {
        if (tid == (i & 1023)) {
            int s = i >> 10;
            bc[0] = obx[s][0]; bc[1] = obx[s][1]; bc[2] = obx[s][2]; bc[3] = obx[s][3];
        }
        __syncthreads();
        if (keepA[i]) {
            float ax1 = bc[0], ay1 = bc[1], ax2 = bc[2], ay2 = bc[3];
            float areaA = (ax2 - ax1) * (ay2 - ay1);
            #pragma unroll
            for (int s = 0; s < 3; ++s) {
                int p = tid + s * 1024;
                if (p > i && p < KALL && keepA[p]) {
                    float bx1 = obx[s][0], by1 = obx[s][1], bx2 = obx[s][2], by2 = obx[s][3];
                    float areaB = (bx2 - bx1) * (by2 - by1);
                    float ix1 = fmaxf(ax1, bx1), iy1 = fmaxf(ay1, by1);
                    float ix2 = fminf(ax2, bx2), iy2 = fminf(ay2, by2);
                    float iw = fmaxf(ix2 - ix1, 0.0f), ih = fmaxf(iy2 - iy1, 0.0f);
                    float inter = iw * ih;
                    float iou = inter / (areaA + areaB - inter + 1e-7f);
                    if (iou > 0.6f) keepA[p] = 0;
                }
            }
        }
        __syncthreads();
    }

    // final = keep ? score : 0; pads -> -1
    for (int p = tid; p < 4096; p += 1024)
        key[p] = (p < KALL) ? (keepA[p] ? key[p] : 0.0f) : -1.0f;
    __syncthreads();

    // top-100 detections
    for (int t = 0; t < NDETS; ++t) {
        float best = -2.0f; int bi = 0;
        for (int p = tid; p < 4096; p += 1024) { if (key[p] > best) { best = key[p]; bi = p; } }
        rs[tid] = best; ri[tid] = bi; __syncthreads();
        for (int st = 512; st > 0; st >>= 1) {
            if (tid < st && rs[tid + st] > rs[tid]) { rs[tid] = rs[tid + st]; ri[tid] = ri[tid + st]; }
            __syncthreads();
        }
        if (tid == 0) {
            int p = ri[0];
            float scv = rs[0] > 0.0f ? rs[0] : 0.0f;
            int jj = sidx[p];
            float* o = outF + ((size_t)n * NDETS + t) * 5;
            o[0] = boxes[((size_t)n * KALL + jj) * 4 + 0];
            o[1] = boxes[((size_t)n * KALL + jj) * 4 + 1];
            o[2] = boxes[((size_t)n * KALL + jj) * 4 + 2];
            o[3] = boxes[((size_t)n * KALL + jj) * 4 + 3];
            o[4] = scv;
            outLab[n * NDETS + t] = labs[n * KALL + jj];
            key[p] = -2.0f;
        }
        __syncthreads();
    }
}

// ---------------------------------------------------------------------------
extern "C" void kernel_launch(void* const* d_in, const int* in_sizes, int n_in,
                              void* d_out, int out_size, void* d_ws, size_t ws_size,
                              hipStream_t stream) {
    static const int   HS[5]  = {100, 50, 25, 13, 7};
    static const int   WSr[5] = {136, 68, 34, 17, 9};
    static const float STR[5] = {8.f, 16.f, 32.f, 64.f, 128.f};

    // Input indices: top-level insertion order; pytree dicts flatten by sorted keys.
    // cls_params: convs[0..2]{b,beta,g,w}, final_b, final_w
    // reg_params: convs[0..2]{b,beta,g,w}, ctr_b, ctr_w, reg_b, reg_w
    const float* feat[5];
    for (int l = 0; l < 5; ++l) feat[l] = (const float*)d_in[l];
    const int clsB[3]    = {5, 9, 13},  clsBeta[3] = {6, 10, 14};
    const int clsG[3]    = {7, 11, 15}, clsW[3]    = {8, 12, 16};
    const float* clsFinB = (const float*)d_in[17];
    const float* clsFinW = (const float*)d_in[18];
    const int regB[3]    = {19, 23, 27}, regBeta[3] = {20, 24, 28};
    const int regG[3]    = {21, 25, 29}, regW[3]    = {22, 26, 30};
    const float* ctrBp   = (const float*)d_in[31];
    const float* ctrWp   = (const float*)d_in[32];
    const float* regFinB = (const float*)d_in[33];
    const float* regFinW = (const float*)d_in[34];

    // Workspace bump allocator
    char* ws = (char*)d_ws;
    size_t off = 0;
    auto alloc = [&](size_t bytes) -> char* {
        char* p = ws + off;
        off += (bytes + 255) & ~(size_t)255;
        return p;
    };

    const size_t maxHW = 13600;
    _Float16* wf16[6];
    for (int i = 0; i < 6; ++i) wf16[i] = (_Float16*)alloc((size_t)CIN * KTOT * 2);
    _Float16* Fh = (_Float16*)alloc((size_t)NB * CIN * maxHW * 2);   // f16 feature
    _Float16* Ph = (_Float16*)alloc((size_t)NB * CIN * maxHW * 2);   // f16 activations
    float*    Q  = (float*)alloc((size_t)NB * CIN * maxHW * 4);      // f32 conv out
    float *clsO[5], *regO[5], *ctrO[5], *scO[5];
    for (int l = 0; l < 5; ++l) {
        int HW = HS[l] * WSr[l];
        clsO[l] = (float*)alloc((size_t)NB * NCLS * HW * 4);
        regO[l] = (float*)alloc((size_t)NB * 4 * HW * 4);
        ctrO[l] = (float*)alloc((size_t)NB * HW * 4);
        scO[l]  = (float*)alloc((size_t)NB * HW * NCLS * 4);
    }
    float* meanB = (float*)alloc(NB * 16 * 4);
    float* rstdB = (float*)alloc(NB * 16 * 4);
    float* candS = (float*)alloc((size_t)NB * KALL * 4);
    int*   candI = (int*)alloc((size_t)NB * KALL * 4);
    float* boxes = (float*)alloc((size_t)NB * KALL * 4 * 4);
    float* scC   = (float*)alloc((size_t)NB * KALL * 4);
    int*   labs  = (int*)alloc((size_t)NB * KALL * 4);

    // Convert trunk conv weights to f16 with (tap,cin) K-order
    {
        int nE = CIN * KTOT;
        int gsz = (nE + 255) / 256;
        for (int j = 0; j < 3; ++j) {
            cvt_w<<<gsz, 256, 0, stream>>>((const float*)d_in[clsW[j]], wf16[j], nE);
            cvt_w<<<gsz, 256, 0, stream>>>((const float*)d_in[regW[j]], wf16[3 + j], nE);
        }
    }

    for (int l = 0; l < 5; ++l) {
        int H = HS[l], Wd = WSr[l], HW = H * Wd;
        dim3 cgrid((unsigned)(((H + 7) / 8) * ((Wd + 7) / 8)), 4, NB);
        int egrid = (NB * CIN * HW + 255) / 256;

        // feature -> f16 once per level
        cvt_f16<<<egrid, 256, 0, stream>>>(feat[l], Fh, NB * CIN * HW);

        // --- classification trunk ---
        const _Float16* cur = Fh;
        for (int j = 0; j < 3; ++j) {
            conv3x3_wmma<<<cgrid, 128, 0, stream>>>(cur, wf16[j], (const float*)d_in[clsB[j]], Q, H, Wd);
            gn_stats<<<NB * 16, 256, 0, stream>>>(Q, meanB, rstdB, HW);
            gn_norm<<<egrid, 256, 0, stream>>>(Q, meanB, rstdB,
                                               (const float*)d_in[clsG[j]],
                                               (const float*)d_in[clsBeta[j]], Ph, HW);
            cur = Ph;
        }
        conv3x3_direct<<<(NB * NCLS * HW + 255) / 256, 256, 0, stream>>>(
            Ph, clsFinW, clsFinB, clsO[l], NCLS, H, Wd, 0);

        // --- regression trunk ---
        cur = Fh;
        for (int j = 0; j < 3; ++j) {
            conv3x3_wmma<<<cgrid, 128, 0, stream>>>(cur, wf16[3 + j], (const float*)d_in[regB[j]], Q, H, Wd);
            gn_stats<<<NB * 16, 256, 0, stream>>>(Q, meanB, rstdB, HW);
            gn_norm<<<egrid, 256, 0, stream>>>(Q, meanB, rstdB,
                                               (const float*)d_in[regG[j]],
                                               (const float*)d_in[regBeta[j]], Ph, HW);
            cur = Ph;
        }
        conv3x3_direct<<<(NB * 4 * HW + 255) / 256, 256, 0, stream>>>(
            Ph, regFinW, regFinB, regO[l], 4, H, Wd, 1);
        conv3x3_direct<<<(NB * 1 * HW + 255) / 256, 256, 0, stream>>>(
            Ph, ctrWp, ctrBp, ctrO[l], 1, H, Wd, 0);

        // --- scores + per-level top-500 + decode ---
        score_kernel<<<(NB * HW + 255) / 256, 256, 0, stream>>>(clsO[l], ctrO[l], scO[l], HW);
        topk_kernel<<<NB, 1024, 0, stream>>>(scO[l], HW * NCLS, candS, candI, l * KLVL);
        decode_kernel<<<(NB * KLVL + 255) / 256, 256, 0, stream>>>(
            candS, candI, regO[l], boxes, scC, labs, Wd, HW, STR[l], l * KLVL);
    }

    float* outF   = (float*)d_out;
    int*   outLab = (int*)(outF + (size_t)NB * NDETS * 5);
    nms_kernel<<<NB, 1024, 0, stream>>>(boxes, scC, labs, outF, outLab);

    (void)in_sizes; (void)n_in; (void)out_size; (void)ws_size;
}
